// ConvMultiHeadAttention_80865644249882
// MI455X (gfx1250) — compile-verified, matrix operand order checked by reading, not run on hardware
//
#include <hip/hip_runtime.h>
#include <hip/hip_bf16.h>
#include <math.h>

// ---------------------------------------------------------------------------
// ConvMultiHeadAttention for MI455X (gfx1250, wave32, WMMA bf16 16x16x32).
//
// Pipeline (all on `stream`):
//   pack_weights   : Wq/Wk/Wv/Wo f32 -> bf16 packed in WMMA A-fragment order
//   conv_wmma<f32in,bf16out> x3 : q/k/v 3x3 conv (implicit GEMM, WMMA) -> Pq/Pk/Pv
//   transpose_v    : Pv -> PvT [b,h,d,seq] for contiguous B-fragments in P@V
//   attn_wmma      : scores = Q@K^T (WMMA), softmax (LDS), out = P@V (WMMA)
//   conv_wmma<bf16in,f32out> : output conv with Wo/bo -> d_out (f32)
//
// Conv GEMM K ordering: K = tap*64 + ic  (tap = ky*3+kx).  A 32-wide K-tile
// always lies inside one tap (tiles are 64-contained), so the shifted-window
// address and bounds check are per-(thread,ktile) scalars, and ic = K & 63.
//
// Workspace layout (assumes ws_size >= ~82 MiB):
//   [0]       Wpk   : 4*18*4*512 bf16 packed weight fragments (288 KiB)
//   [1 MiB]   Pq    : 512*64*256 bf16 (16 MiB)
//   [17 MiB]  Pk    : 16 MiB
//   [33 MiB]  Pv    : 16 MiB
//   [49 MiB]  PvT   : 16 MiB
//   [65 MiB]  Attn  : 16 MiB
// ---------------------------------------------------------------------------

typedef __attribute__((ext_vector_type(16))) __bf16 v16bf;
typedef __attribute__((ext_vector_type(8)))  __bf16 v8bf;
typedef __attribute__((ext_vector_type(8)))  float  v8f;

#define SEQ    256
#define CCH    64
#define HW     256           // 16*16 pixels
#define DHEAD  2048          // d_k * H * W = 8*256
#define FRAMES 512           // bs*seq
#define KDIM   576           // 9*64
#define KT_CONV 18           // 576/32
#define ATTN_SCALE 0.02209708691f   // 1/sqrt(2048)

__device__ __forceinline__ v8f wmma_bf16(v16bf a, v16bf b, v8f c) {
    // D = A(16x32) * B(32x16) + C, f32 accumulate
    return __builtin_amdgcn_wmma_f32_16x16x32_bf16(
        /*neg_a=*/false, a, /*neg_b=*/false, b,
        /*c_mod=*/(short)0, c, /*reuse_a=*/false, /*reuse_b=*/false);
}

// 16-bit operand per-lane K map (ISA 7.12.2):
//   elements 0..7  -> K = e      + 8*khalf
//   elements 8..15 -> K = e + 8  + 8*khalf
// i.e. two contiguous 8-element (16B) chunks at +8*khalf and +16+8*khalf.
__device__ __forceinline__ v16bf load_frag(const __bf16* __restrict__ p, int khalf) {
    v8bf lo = *(const v8bf*)(p + 8 * khalf);
    v8bf hi = *(const v8bf*)(p + 16 + 8 * khalf);
    return __builtin_shufflevector(lo, hi, 0,1,2,3,4,5,6,7,8,9,10,11,12,13,14,15);
}

// ---------------------------------------------------------------------------
// Kernel 1: pack conv weights into per-lane WMMA A-fragment order.
// blockIdx.x = c*72 + kt*4 + ot   (c: conv id 0..3, kt: K-tile 0..17, ot: oc-tile 0..3)
// threadIdx.x = l*16 + e          (l: lane 0..31, e: element 0..15)
// K ordering: K = tap*64 + ic
// ---------------------------------------------------------------------------
__global__ void pack_weights(const float* __restrict__ Wq, const float* __restrict__ Wk,
                             const float* __restrict__ Wv, const float* __restrict__ Wo,
                             __bf16* __restrict__ Wpk) {
    const int blk = blockIdx.x;                 // 0..287
    const int c   = blk / 72;
    const int rem = blk - c * 72;
    const int kt  = rem >> 2;
    const int ot  = rem & 3;
    const int t   = threadIdx.x;                // 0..511
    const int e   = t & 15;
    const int l   = t >> 4;
    const int khalf = l >> 4;
    const int lm    = l & 15;

    const int K   = kt * 32 + ((e < 8) ? (e + 8 * khalf) : (e + 8 + 8 * khalf));
    const int m   = ot * 16 + lm;               // output channel row
    const int ic  = K & 63;
    const int tap = K >> 6;                     // ky*3+kx

    const float* W = (c == 0) ? Wq : (c == 1) ? Wk : (c == 2) ? Wv : Wo;
    Wpk[blk * 512 + l * 16 + e] = (__bf16)W[(m * 64 + ic) * 9 + tap];
}

// ---------------------------------------------------------------------------
// Kernel 2: implicit-GEMM 3x3 conv via WMMA.  One frame per 256-thread block.
//   D[oc 64 x px 256] = W[64 x 576] * im2col[576 x 256] + bias
// Per K-tile: all 256 threads cooperatively stage the 32x256 im2col slab in
// LDS *in fragment order* bt[nt][lane][16]; then wave w (M-tile w&3, N-tiles
// (w>>2)*8..+8) consumes it with one 32B ds load per WMMA.
// ---------------------------------------------------------------------------
template <bool IN_F32, bool OUT_F32>
__global__ void conv_wmma(const void* __restrict__ inp,
                          const __bf16* __restrict__ wpk,   // this conv's 72 fragment blocks
                          const float* __restrict__ bias,
                          void* __restrict__ outp) {
    __shared__ __align__(16) __bf16 fr[CCH * HW];     // 32 KiB frame (bf16)
    __shared__ __align__(16) __bf16 bt[16 * 32 * 16]; // 16 KiB B-tile, fragment order

    const int frame = blockIdx.x;
    const int tid   = threadIdx.x;                    // 256 threads, 8 waves

    // ---- stage frame into LDS (convert f32 -> bf16 on the way in) ----
    if (IN_F32) {
        const float4* src = (const float4*)((const float*)inp + (size_t)frame * CCH * HW);
#pragma unroll
        for (int i = 0; i < 16; ++i) {
            float4 v = src[tid + 256 * i];
            int o = 4 * (tid + 256 * i);
            fr[o + 0] = (__bf16)v.x; fr[o + 1] = (__bf16)v.y;
            fr[o + 2] = (__bf16)v.z; fr[o + 3] = (__bf16)v.w;
        }
    } else {
        const v8bf* src = (const v8bf*)((const __bf16*)inp + (size_t)frame * CCH * HW);
#pragma unroll
        for (int i = 0; i < 8; ++i) {
            *(v8bf*)(fr + 8 * (tid + 256 * i)) = src[tid + 256 * i];
        }
    }

    const int lane  = tid & 31;
    const int w     = tid >> 5;
    const int khalf = lane >> 4;
    const int lm    = lane & 15;
    const int mtile = w & 3;
    const int ocb   = mtile * 16;
    const int nt0   = (w >> 2) * 8;

    // Builder role: thread = pixel column px = tid (nt = tid>>4, col-in-tile = tid&15).
    const int py = tid >> 4, pxx = tid & 15;

    v8f acc[8];
#pragma unroll
    for (int nn = 0; nn < 8; ++nn)
        acc[nn] = (v8f){0.f, 0.f, 0.f, 0.f, 0.f, 0.f, 0.f, 0.f};

#pragma unroll
    for (int kt = 0; kt < KT_CONV; ++kt) {
        __syncthreads();   // previous tile fully consumed / frame staged

        // ---- cooperative B-tile build (fragment order) ----
        {
            const int tap = kt >> 1;                 // constant tap per K-tile
            const int ky  = tap / 3, kx = tap - ky * 3;
            const int yy  = py + ky - 1, xx = pxx + kx - 1;
            const bool inb = ((unsigned)yy < 16u) && ((unsigned)xx < 16u);
            const int  off = yy * 16 + xx;           // shifted pixel, fixed per tile
            const int  icb = (kt & 1) * 32;          // ic base of this 32-wide K-tile
#pragma unroll
            for (int kh = 0; kh < 2; ++kh) {
                v16bf tmp = (v16bf)((__bf16)0.f);
                if (inb) {
                    const __bf16* g = fr + (icb + 8 * kh) * HW + off;
#pragma unroll
                    for (int e = 0; e < 16; ++e) {
                        const int ice = (e < 8) ? e : (e + 8);   // ic offset per element
                        tmp[e] = g[ice * HW];
                    }
                }
                // bt[(nt*32 + kh*16 + lm)*16 .. +15] : nt = tid>>4, lm = tid&15
                *(v16bf*)(bt + ((tid >> 4) * 32 + kh * 16 + (tid & 15)) * 16) = tmp;
            }
        }
        __syncthreads();   // tile visible to all waves

        // ---- consume: one A-fragment load + 8 WMMAs per wave ----
        const v16bf af = *(const v16bf*)(wpk + (size_t)(kt * 4 + mtile) * 512 + lane * 16);
#pragma unroll
        for (int nn = 0; nn < 8; ++nn) {
            const v16bf bfr = *(const v16bf*)(bt + ((nt0 + nn) * 32 + lane) * 16);
            acc[nn] = wmma_bf16(af, bfr, acc[nn]);
        }
    }

    // ---- C/D layout: lane col n=lm, row oc = ocb + 8*khalf + r ----
#pragma unroll
    for (int nn = 0; nn < 8; ++nn) {
        const int px = (nt0 + nn) * 16 + lm;
#pragma unroll
        for (int r = 0; r < 8; ++r) {
            const int oc  = ocb + 8 * khalf + r;
            const float v = acc[nn][r] + bias[oc];
            const size_t o = (size_t)frame * CCH * HW + (size_t)oc * HW + px;
            if (OUT_F32) ((float*)outp)[o]  = v;
            else         ((__bf16*)outp)[o] = (__bf16)v;
        }
    }
}

// ---------------------------------------------------------------------------
// Kernel 3: Pv [frame][oc][px]  ->  PvT [b][h][d=2048][seq=256]
// ---------------------------------------------------------------------------
__global__ void transpose_v(const __bf16* __restrict__ Pv, __bf16* __restrict__ PvT) {
    const int idx = blockIdx.x * 256 + threadIdx.x;   // 0 .. 2*8*2048*256-1
    const int s  = idx & 255;
    const int d  = (idx >> 8) & 2047;
    const int h  = (idx >> 19) & 7;
    const int b  = idx >> 22;
    PvT[idx] = Pv[(size_t)(b * 256 + s) * (CCH * HW) + (size_t)(h * 8 + (d >> 8)) * HW + (d & 255)];
}

// ---------------------------------------------------------------------------
// Kernel 4: attention.  One block = (b, h, 32 query rows); 256 threads / 8 waves.
//   Phase 1: S[32x256] = Q[32x2048] @ K^T       (WMMA, 2048 wmma/block)
//   Phase 2: row softmax in LDS -> Pm bf16
//   Phase 3: O[32x2048] = Pm @ V                 (WMMA, 2048 wmma/block)
// ---------------------------------------------------------------------------
__global__ void attn_wmma(const __bf16* __restrict__ Pq, const __bf16* __restrict__ Pk,
                          const __bf16* __restrict__ PvT, __bf16* __restrict__ Attn) {
    __shared__ __align__(16) float  S[32 * 256];     // 32 KiB scores
    __shared__ __align__(16) __bf16 Pm[32 * 256];    // 16 KiB probabilities
    __shared__ float red[32 * 8];
    __shared__ float rowstat[32];

    const int tid   = threadIdx.x;
    const int lane  = tid & 31;
    const int w     = tid >> 5;
    const int khalf = lane >> 4;
    const int lm    = lane & 15;

    const int b     = blockIdx.x >> 6;
    const int h     = (blockIdx.x >> 3) & 7;
    const int qb    = blockIdx.x & 7;
    const int qrow0 = qb * 32;
    const int mtile = w >> 2;                        // 0 or 1

    // ---------------- Phase 1: scores = Q @ K^T ----------------
    {
        const int nt0 = (w & 3) * 4;
        v8f acc[4];
#pragma unroll
        for (int j = 0; j < 4; ++j) acc[j] = (v8f){0.f,0.f,0.f,0.f,0.f,0.f,0.f,0.f};

        // A lane row = query row; B lane column = key row (both K-contiguous).
        const __bf16* qrow = Pq + ((size_t)(b * 256 + qrow0 + mtile * 16 + lm) * CCH + h * 8) * HW;
        const __bf16* krow[4];
#pragma unroll
        for (int j = 0; j < 4; ++j) {
            const int key = (nt0 + j) * 16 + lm;
            krow[j] = Pk + ((size_t)(b * 256 + key) * CCH + h * 8) * HW;
        }

        for (int kt = 0; kt < 64; ++kt) {
            __builtin_prefetch(qrow + kt * 32 + 256, 0, 0);     // global_prefetch_b8
            v16bf afr = load_frag(qrow + kt * 32, khalf);
#pragma unroll
            for (int j = 0; j < 4; ++j) {
                v16bf bfr = load_frag(krow[j] + kt * 32, khalf);
                acc[j] = wmma_bf16(afr, bfr, acc[j]);
            }
        }
        // C layout: lane col n=lm, row m = mtile*16 + 8*khalf + r
#pragma unroll
        for (int j = 0; j < 4; ++j)
#pragma unroll
            for (int r = 0; r < 8; ++r)
                S[(mtile * 16 + 8 * khalf + r) * 256 + (nt0 + j) * 16 + lm] = acc[j][r] * ATTN_SCALE;
    }
    __syncthreads();

    // ---------------- Phase 2: softmax over rows ----------------
    {
        const int r  = tid >> 3;          // 32 rows, 8 threads each
        const int c0 = (tid & 7) * 32;
        float mx = -3.4e38f;
        for (int j = 0; j < 32; ++j) mx = fmaxf(mx, S[r * 256 + c0 + j]);
        red[r * 8 + (tid & 7)] = mx;
        __syncthreads();
        if ((tid & 7) == 0) {
            float m = red[r * 8];
#pragma unroll
            for (int k2 = 1; k2 < 8; ++k2) m = fmaxf(m, red[r * 8 + k2]);
            rowstat[r] = m;
        }
        __syncthreads();
        const float m = rowstat[r];
        float s = 0.f;
        for (int j = 0; j < 32; ++j) {
            const float e = __expf(S[r * 256 + c0 + j] - m);
            S[r * 256 + c0 + j] = e;
            s += e;
        }
        __syncthreads();
        red[r * 8 + (tid & 7)] = s;
        __syncthreads();
        if ((tid & 7) == 0) {
            float t = 0.f;
#pragma unroll
            for (int k2 = 0; k2 < 8; ++k2) t += red[r * 8 + k2];
            rowstat[r] = t;
        }
        __syncthreads();
        const float inv = 1.0f / rowstat[r];
        for (int j = 0; j < 32; ++j)
            Pm[r * 256 + c0 + j] = (__bf16)(S[r * 256 + c0 + j] * inv);
    }
    __syncthreads();

    // ---------------- Phase 3: out = Pm @ V ----------------
    {
        // Cache the 8 A fragments (K = 256 keys) for this wave's query tile.
        v16bf af[8];
#pragma unroll
        for (int kt = 0; kt < 8; ++kt)
            af[kt] = load_frag(Pm + (mtile * 16 + lm) * 256 + kt * 32, khalf);

        const size_t vbase = (size_t)(b * 8 + h) * DHEAD * SEQ;
#pragma unroll 1
        for (int i = 0; i < 32; ++i) {
            const int dt = (w & 3) + 4 * i;           // d-tile 0..127
            const int d  = dt * 16 + lm;              // B column = d index
            const __bf16* vrow = PvT + vbase + (size_t)d * SEQ;

            v8f acc = {0.f, 0.f, 0.f, 0.f, 0.f, 0.f, 0.f, 0.f};
#pragma unroll
            for (int kt = 0; kt < 8; ++kt) {
                v16bf bfr = load_frag(vrow + kt * 32, khalf);
                acc = wmma_bf16(af[kt], bfr, acc);
            }

            const int dk = d >> 8, px = d & 255;
#pragma unroll
            for (int r = 0; r < 8; ++r) {
                const int qrow = qrow0 + mtile * 16 + 8 * khalf + r;
                Attn[((size_t)(b * 256 + qrow) * CCH + h * 8 + dk) * HW + px] = (__bf16)acc[r];
            }
        }
    }
}

// ---------------------------------------------------------------------------
extern "C" void kernel_launch(void* const* d_in, const int* in_sizes, int n_in,
                              void* d_out, int out_size, void* d_ws, size_t ws_size,
                              hipStream_t stream) {
    const float* q  = (const float*)d_in[0];
    const float* k  = (const float*)d_in[1];
    const float* v  = (const float*)d_in[2];
    const float* Wq = (const float*)d_in[3];
    const float* bq = (const float*)d_in[4];
    const float* Wk = (const float*)d_in[5];
    const float* bk = (const float*)d_in[6];
    const float* Wv = (const float*)d_in[7];
    const float* bv = (const float*)d_in[8];
    const float* Wo = (const float*)d_in[9];
    const float* bo = (const float*)d_in[10];

    char* ws = (char*)d_ws;
    const size_t MiB = 1024ull * 1024ull;
    __bf16* Wpk  = (__bf16*)(ws);                 // 288 KiB
    __bf16* Pq   = (__bf16*)(ws + 1  * MiB);      // 16 MiB each
    __bf16* Pk   = (__bf16*)(ws + 17 * MiB);
    __bf16* Pv   = (__bf16*)(ws + 33 * MiB);
    __bf16* PvT  = (__bf16*)(ws + 49 * MiB);
    __bf16* At   = (__bf16*)(ws + 65 * MiB);

    // 1) pack weights (fragment layout, K = tap*64 + ic)
    pack_weights<<<288, 512, 0, stream>>>(Wq, Wk, Wv, Wo, Wpk);

    // 2) q/k/v projections (WMMA implicit-GEMM conv, f32 in -> bf16 out)
    conv_wmma<true, false><<<FRAMES, 256, 0, stream>>>(q, Wpk + 0 * 72 * 512, bq, Pq);
    conv_wmma<true, false><<<FRAMES, 256, 0, stream>>>(k, Wpk + 1 * 72 * 512, bk, Pk);
    conv_wmma<true, false><<<FRAMES, 256, 0, stream>>>(v, Wpk + 2 * 72 * 512, bv, Pv);

    // 3) V transpose for contiguous B-fragments in P@V
    transpose_v<<<(2 * 8 * DHEAD * SEQ) / 256, 256, 0, stream>>>(Pv, PvT);

    // 4) attention (2 b * 8 h * 8 q-blocks = 128 blocks)
    attn_wmma<<<128, 256, 0, stream>>>(Pq, Pk, PvT, At);

    // 5) output conv (bf16 in -> f32 out) writes d_out
    conv_wmma<false, true><<<FRAMES, 256, 0, stream>>>(At, Wpk + 3 * 72 * 512, bo, d_out);
}
